// ColorizationNet_52948356825890
// MI455X (gfx1250) — matrix operand order repaired and checked by reading, not compiled, not run on hardware
//
#include <hip/hip_runtime.h>

typedef __attribute__((ext_vector_type(16))) __bf16 v16bf;
typedef __attribute__((ext_vector_type(8)))  float  v8f;

#define IMG   256
#define H1N   304
#define H1P   320   // padded K for gemm2
#define H2N   176
#define H2P   192   // padded K for gemm3
#define OUTN  48
#define NCHUNK 4096

__device__ __forceinline__ unsigned short f2bf(float f) {
  unsigned int u = __builtin_bit_cast(unsigned int, f);
  u = (u + 0x7FFFu + ((u >> 16) & 1u)) >> 16;   // round-to-nearest-even
  return (unsigned short)u;
}

// ---------------- conv backbone (direct conv + fused relu/maxpool) ----------------

__global__ void conv1_kernel(const float* __restrict__ x, const float* __restrict__ w,
                             const float* __restrict__ b, float* __restrict__ p1) {
  int idx = blockIdx.x * 256 + threadIdx.x;          // 8*128*128
  if (idx >= 8 * 128 * 128) return;
  int oc = idx >> 14, y = (idx >> 7) & 127, xo = idx & 127;
  float wv[9];
#pragma unroll
  for (int t = 0; t < 9; ++t) wv[t] = w[oc * 9 + t];
  float bias = b[oc], m = 0.f;
#pragma unroll
  for (int dy = 0; dy < 2; ++dy)
#pragma unroll
    for (int dx = 0; dx < 2; ++dx) {
      int cy = 2 * y + dy, cx = 2 * xo + dx;
      float acc = bias;
#pragma unroll
      for (int r = 0; r < 3; ++r) {
        int iy = cy + r - 1;
        if (iy < 0 || iy >= 256) continue;
#pragma unroll
        for (int s = 0; s < 3; ++s) {
          int ix = cx + s - 1;
          if (ix < 0 || ix >= 256) continue;
          acc = fmaf(x[iy * 256 + ix], wv[r * 3 + s], acc);
        }
      }
      m = fmaxf(m, acc);   // relu folded into maxpool (all candidates clamped at 0)
    }
  p1[idx] = m;
}

__global__ void conv2_kernel(const float* __restrict__ p1, const float* __restrict__ w,
                             const float* __restrict__ b, float* __restrict__ p2) {
  int idx = blockIdx.x * 256 + threadIdx.x;          // 16*64*64
  if (idx >= 16 * 64 * 64) return;
  int oc = idx >> 12, y = (idx >> 6) & 63, xo = idx & 63;
  float bias = b[oc], m = 0.f;
#pragma unroll
  for (int dy = 0; dy < 2; ++dy)
#pragma unroll
    for (int dx = 0; dx < 2; ++dx) {
      int cy = 2 * y + dy, cx = 2 * xo + dx;
      float acc = bias;
      for (int ic = 0; ic < 8; ++ic) {
        const float* xin = p1 + ic * 128 * 128;
        const float* wk = w + (oc * 8 + ic) * 9;
#pragma unroll
        for (int r = 0; r < 3; ++r) {
          int iy = cy + r - 1;
          if (iy < 0 || iy >= 128) continue;
#pragma unroll
          for (int s = 0; s < 3; ++s) {
            int ix = cx + s - 1;
            if (ix < 0 || ix >= 128) continue;
            acc = fmaf(xin[iy * 128 + ix], wk[r * 3 + s], acc);
          }
        }
      }
      m = fmaxf(m, acc);
    }
  p2[idx] = m;
}

__global__ void conv3_kernel(const float* __restrict__ p2, const float* __restrict__ w,
                             const float* __restrict__ b, float* __restrict__ p3) {
  int idx = blockIdx.x * 256 + threadIdx.x;          // 32*32*32
  if (idx >= 32 * 32 * 32) return;
  int oc = idx >> 10, y = (idx >> 5) & 31, xo = idx & 31;
  float bias = b[oc], m = 0.f;
#pragma unroll
  for (int dy = 0; dy < 2; ++dy)
#pragma unroll
    for (int dx = 0; dx < 2; ++dx) {
      int cy = 2 * y + dy, cx = 2 * xo + dx;
      float acc = bias;
      for (int ic = 0; ic < 16; ++ic) {
        const float* xin = p2 + ic * 64 * 64;
        const float* wk = w + (oc * 16 + ic) * 9;
#pragma unroll
        for (int r = 0; r < 3; ++r) {
          int iy = cy + r - 1;
          if (iy < 0 || iy >= 64) continue;
#pragma unroll
          for (int s = 0; s < 3; ++s) {
            int ix = cx + s - 1;
            if (ix < 0 || ix >= 64) continue;
            acc = fmaf(xin[iy * 64 + ix], wk[r * 3 + s], acc);
          }
        }
      }
      m = fmaxf(m, acc);
    }
  p3[idx] = m;
}

// ------------- shared matvec: s[k] = b1[k] + w1[k, :32768] . x_conv (f32, one w1 pass) -------------

__global__ void matvec_kernel(const float* __restrict__ w1, const float* __restrict__ b1,
                              const float* __restrict__ xc, float* __restrict__ s) {
  __shared__ float red[256];
  int k = blockIdx.x;                                // 0..303
  const float2* row = (const float2*)(w1 + (size_t)k * 32786);
  const float2* xv  = (const float2*)xc;
  float acc = 0.f;
  for (int t = threadIdx.x; t < 16384; t += 256) {
    float2 a = row[t], bb = xv[t];
    acc = fmaf(a.x, bb.x, acc);
    acc = fmaf(a.y, bb.y, acc);
  }
  red[threadIdx.x] = acc;
  __syncthreads();
  for (int off = 128; off > 0; off >>= 1) {
    if (threadIdx.x < off) red[threadIdx.x] += red[threadIdx.x + off];
    __syncthreads();
  }
  if (threadIdx.x == 0) s[k] = red[0] + b1[k];
}

// ------------- h1 = relu(s + pos/chunk rank-18 update), stored bf16 padded [4096 x 320] -------------

__global__ void h1_kernel(const float* __restrict__ x, const float* __restrict__ w1,
                          const float* __restrict__ s, unsigned short* __restrict__ h1b) {
  int idx = blockIdx.x * 256 + threadIdx.x;          // 4096*320
  if (idx >= NCHUNK * H1P) return;
  int c = idx / H1P;
  int k = idx - c * H1P;
  if (k >= H1N) { h1b[idx] = 0; return; }            // K padding
  int i = c >> 6, j = c & 63;
  const float* wr = w1 + (size_t)k * 32786 + 32768;  // [pos0, pos1, chunk 0..15]
  float acc = s[k];
  acc = fmaf((float)(i * 4), wr[0], acc);
  acc = fmaf((float)(j * 4), wr[1], acc);
  const float* xp = x + (i * 4) * IMG + (j * 4);
#pragma unroll
  for (int r = 0; r < 4; ++r)
#pragma unroll
    for (int t = 0; t < 4; ++t)
      acc = fmaf(xp[r * IMG + t], wr[2 + r * 4 + t], acc);
  h1b[idx] = f2bf(fmaxf(acc, 0.f));
}

// ------------- weight repack f32 -> bf16, transposed+padded for WMMA B-fragments -------------

__global__ void prepB2_kernel(const float* __restrict__ w2, unsigned short* __restrict__ B2) {
  int idx = blockIdx.x * 256 + threadIdx.x;          // 320*176
  if (idx >= H1P * H2N) return;
  int n = idx / H1P, k = idx - n * H1P;
  B2[idx] = f2bf(k < H1N ? w2[n * H1N + k] : 0.f);
}

__global__ void prepB3_kernel(const float* __restrict__ w3, unsigned short* __restrict__ B3) {
  int idx = blockIdx.x * 256 + threadIdx.x;          // 192*48
  if (idx >= H2P * OUTN) return;
  int n = idx / H2P, k = idx - n * H2P;
  B3[idx] = f2bf(k < H2N ? w3[n * H2N + k] : 0.f);
}

__global__ void padh2_kernel(unsigned short* __restrict__ h2b) {
  int idx = blockIdx.x * 256 + threadIdx.x;          // 4096*16 pad region
  int m = idx >> 4, n = H2N + (idx & 15);
  h2b[m * H2P + n] = 0;
}

// ------------- WMMA GEMM2: h2[4096,176] = relu(h1[4096,320] @ B2[320,176] + b2) -------------

__global__ void __launch_bounds__(128) gemm2_kernel(const unsigned short* __restrict__ h1b,
                                                    const unsigned short* __restrict__ B2,
                                                    const float* __restrict__ b2,
                                                    unsigned short* __restrict__ h2b) {
  const int lane = threadIdx.x & 31;
  const int wave = threadIdx.x >> 5;
  const int tile = blockIdx.x * 4 + wave;            // 704*4 == 2816 == 256*11 exactly
  const int tM = tile / 11;
  const int tN = tile - tM * 11;
  const int half = lane >> 4;
  const int mrow = tM * 16 + (lane & 15);
  union Frag { v16bf v; uint4 q[2]; };
  v8f acc = {};
#pragma unroll
  for (int k0 = 0; k0 < H1P; k0 += 32) {
    Frag a, b;
    // A: lane = M row; elems 0-7 <- K[k0+half*8 ..], elems 8-15 <- K[k0+16+half*8 ..]
    const uint4* pa = (const uint4*)(h1b + mrow * H1P + k0 + half * 8);
    a.q[0] = pa[0];
    a.q[1] = pa[2];
    // B: lane = N col; elems 0-7 <- K[k0+half*8 ..], elems 8-15 <- K[k0+16+half*8 ..]
    const uint4* pb = (const uint4*)(B2 + (tN * 16 + (lane & 15)) * H1P + k0 + half * 8);
    b.q[0] = pb[0];
    b.q[1] = pb[2];
    acc = __builtin_amdgcn_wmma_f32_16x16x32_bf16(false, a.v, false, b.v,
                                                  (short)0, acc, false, false);
  }
  const int n = tN * 16 + (lane & 15);
  const float bias = b2[n];
#pragma unroll
  for (int r = 0; r < 8; ++r) {
    const int m = tM * 16 + r + half * 8;
    h2b[m * H2P + n] = f2bf(fmaxf(acc[r] + bias, 0.f));
  }
}

// ------------- WMMA GEMM3: out[4096,48] = sigmoid(h2[4096,192] @ B3[192,48] + b3) -------------

__global__ void __launch_bounds__(128) gemm3_kernel(const unsigned short* __restrict__ h2b,
                                                    const unsigned short* __restrict__ B3,
                                                    const float* __restrict__ b3,
                                                    float* __restrict__ out) {
  const int lane = threadIdx.x & 31;
  const int wave = threadIdx.x >> 5;
  const int tile = blockIdx.x * 4 + wave;            // 192*4 == 768 == 256*3 exactly
  const int tM = tile / 3;
  const int tN = tile - tM * 3;
  const int half = lane >> 4;
  const int mrow = tM * 16 + (lane & 15);
  union Frag { v16bf v; uint4 q[2]; };
  v8f acc = {};
#pragma unroll
  for (int k0 = 0; k0 < H2P; k0 += 32) {
    Frag a, b;
    const uint4* pa = (const uint4*)(h2b + mrow * H2P + k0 + half * 8);
    a.q[0] = pa[0];
    a.q[1] = pa[2];
    const uint4* pb = (const uint4*)(B3 + (tN * 16 + (lane & 15)) * H2P + k0 + half * 8);
    b.q[0] = pb[0];
    b.q[1] = pb[2];
    acc = __builtin_amdgcn_wmma_f32_16x16x32_bf16(false, a.v, false, b.v,
                                                  (short)0, acc, false, false);
  }
  const int n = tN * 16 + (lane & 15);
  const float bias = b3[n];
#pragma unroll
  for (int r = 0; r < 8; ++r) {
    const int m = tM * 16 + r + half * 8;
    float v = acc[r] + bias;
    out[m * OUTN + n] = 1.f / (1.f + __expf(-v));    // flat [4096,48] == reshape(3,256,256)
  }
}

// ----------------------------------- launcher -----------------------------------

extern "C" void kernel_launch(void* const* d_in, const int* in_sizes, int n_in,
                              void* d_out, int out_size, void* d_ws, size_t ws_size,
                              hipStream_t stream) {
  const float* x   = (const float*)d_in[0];
  const float* c1w = (const float*)d_in[1];
  const float* c1b = (const float*)d_in[2];
  const float* c2w = (const float*)d_in[3];
  const float* c2b = (const float*)d_in[4];
  const float* c3w = (const float*)d_in[5];
  const float* c3b = (const float*)d_in[6];
  const float* w1  = (const float*)d_in[7];
  const float* b1  = (const float*)d_in[8];
  const float* w2  = (const float*)d_in[9];
  const float* b2  = (const float*)d_in[10];
  const float* w3  = (const float*)d_in[11];
  const float* b3  = (const float*)d_in[12];
  float* out = (float*)d_out;

  char* ws = (char*)d_ws;
  float* p1 = (float*)(ws + 0);                       //   8*128*128*4 = 524288
  float* p2 = (float*)(ws + 524288);                  //  16* 64* 64*4 = 262144
  float* p3 = (float*)(ws + 786432);                  //  32768*4      = 131072
  float* s  = (float*)(ws + 917504);                  //  304*4 (+pad) =   2048
  unsigned short* h1b = (unsigned short*)(ws + 919552);   // 4096*320*2 = 2621440
  unsigned short* B2  = (unsigned short*)(ws + 3540992);  //  320*176*2 =  112640
  unsigned short* h2b = (unsigned short*)(ws + 3653632);  // 4096*192*2 = 1572864
  unsigned short* B3  = (unsigned short*)(ws + 5226496);  //  192* 48*2 =   18432

  conv1_kernel<<<512, 256, 0, stream>>>(x, c1w, c1b, p1);
  conv2_kernel<<<256, 256, 0, stream>>>(p1, c2w, c2b, p2);
  conv3_kernel<<<128, 256, 0, stream>>>(p2, c3w, c3b, p3);
  matvec_kernel<<<H1N, 256, 0, stream>>>(w1, b1, p3, s);
  prepB2_kernel<<<(H1P * H2N + 255) / 256, 256, 0, stream>>>(w2, B2);
  prepB3_kernel<<<(H2P * OUTN + 255) / 256, 256, 0, stream>>>(w3, B3);
  h1_kernel<<<(NCHUNK * H1P) / 256, 256, 0, stream>>>(x, w1, s, h1b);
  padh2_kernel<<<(NCHUNK * 16) / 256, 256, 0, stream>>>(h2b);
  gemm2_kernel<<<704, 128, 0, stream>>>(h1b, B2, b2, h2b);
  gemm3_kernel<<<192, 128, 0, stream>>>(h2b, B3, b3, out);
}
